// TTS_64029372449173
// MI455X (gfx1250) — compile-verified, matrix operand order checked by reading, not run on hardware
//
#include <hip/hip_runtime.h>

// ---------------- sizes ----------------
#define B_  32
#define T_  512
#define S_  512
#define D_  1024
#define M_  10
#define K_  3072          // 2D (x_t|ctx) + D (h)
#define N4_ 4096          // 4*D gates
#define GRID_BLOCKS 128
#define BLOCK_T 128

typedef __attribute__((ext_vector_type(16))) __bf16       v16bf;
typedef __attribute__((ext_vector_type(8)))  float        v8f;
typedef __attribute__((ext_vector_type(4)))  unsigned int v4u;

union U16bf { v16bf v; v4u u[2]; };

__device__ __forceinline__ unsigned short f2bf(float f) {
  unsigned u = __float_as_uint(f);
  u += 0x7FFFu + ((u >> 16) & 1u);          // round-to-nearest-even
  return (unsigned short)(u >> 16);
}
__device__ __forceinline__ float sigm(float x) { return 1.f / (1.f + __expf(-x)); }
__device__ __forceinline__ float tanh_f(float x) {
  x = fminf(fmaxf(x, -15.f), 15.f);
  float e = __expf(2.f * x);
  return (e - 1.f) / (e + 1.f);
}

// grid-wide sense barrier (all 128 blocks resident -> safe)
__device__ __forceinline__ void gsync(unsigned* cnt, volatile unsigned* gen) {
  __syncthreads();
  if (threadIdx.x == 0) {
    __threadfence();
    unsigned g = *gen;
    if (atomicAdd(cnt, 1u) == GRID_BLOCKS - 1u) {
      *cnt = 0u;
      __threadfence();
      *gen = g + 1u;
    } else {
      while (*gen == g) { __builtin_amdgcn_s_sleep(1); }
    }
    __threadfence();
  }
  __syncthreads();
}

// ---------------- init kernels ----------------
// Wcat bf16 [4096][3072]: row n = [W_ih[n,0:2048] | W_hh[n,0:1024]]
__global__ void tts_init_wcat(const float* __restrict__ W_ih,
                              const float* __restrict__ W_hh,
                              unsigned short* __restrict__ wcat) {
  long idx = (long)blockIdx.x * blockDim.x + threadIdx.x;
  if (idx >= (long)N4_ * K_) return;
  int n = (int)(idx / K_), k = (int)(idx - (long)n * K_);
  float v = (k < 2 * D_) ? W_ih[(long)n * (2 * D_) + k]
                         : W_hh[(long)n * D_ + (k - 2 * D_)];
  wcat[idx] = f2bf(v);
}

// state init: xbuf (x_0 | 0 | 0), cbuf=0, bias=b_ih+b_hh, ksi(double-buf)=0, sync=0
__global__ void tts_init_state(const float* __restrict__ input_h_c,
                               const float* __restrict__ b_ih,
                               const float* __restrict__ b_hh,
                               unsigned short* __restrict__ xbuf,
                               float* __restrict__ cbuf,
                               float* __restrict__ bias,
                               float* __restrict__ ksib,
                               unsigned* __restrict__ syncp) {
  int idx = blockIdx.x * blockDim.x + threadIdx.x;
  if (idx < B_ * K_) {
    int b = idx / K_, col = idx - b * K_;
    xbuf[idx] = (col < D_) ? f2bf(input_h_c[(long)b * T_ * D_ + col]) : (unsigned short)0;
  }
  if (idx < B_ * D_) cbuf[idx] = 0.f;
  if (idx < N4_)     bias[idx] = b_ih[idx] + b_hh[idx];
  if (idx < 2 * B_ * M_) ksib[idx] = 0.f;
  if (idx < 2)       syncp[idx] = 0u;
}

// ---------------- persistent recurrent kernel ----------------
__global__ __launch_bounds__(BLOCK_T, 1) void tts_persistent(
    const float* __restrict__ input_h_c, const float* __restrict__ memory,
    const int* __restrict__ input_lengths, const float* __restrict__ W_g,
    const float* __restrict__ b_g,
    const unsigned short* __restrict__ wcat, const float* __restrict__ bias,
    unsigned short* __restrict__ xbuf, float* __restrict__ cbuf,
    float* __restrict__ ksib, float* __restrict__ part,
    float* __restrict__ wbuf, unsigned* __restrict__ syncp,
    float* __restrict__ out_ctx, float* __restrict__ out_align,
    float* __restrict__ out_term) {
  __shared__ float smem[4 * 256];  // 4 gate tiles 16x16, slot0 reused for h
  __shared__ float sphi[30];
  __shared__ float sk[10], sbi[10], sa[10];

  const int tid  = threadIdx.x;
  const int blk  = blockIdx.x;
  const int wave = tid >> 5;
  const int lane = tid & 31;
  const int half = lane >> 4;     // K-half select per WMMA 16-bit A/B layout
  const int lm   = lane & 15;     // row (A) / col (B) within tile

  const int mtile = blk >> 6;                 // 0..1  (batch tile)
  const int jtile = blk & 63;                 // 0..63 (16-col chunk of D)
  const int m0 = mtile * 16;
  const int j0g = jtile * 16;
  const int n0 = wave * D_ + j0g;             // column in 4D gate space

  unsigned*          cnt = syncp;
  volatile unsigned* gen = syncp + 1;

  for (int t = 0; t < T_; ++t) {
    // ---------- phase A: gates GEMM (WMMA bf16) + LSTM pointwise + phi partials
    {
      const unsigned short* Ab = xbuf + (long)(m0 + lm) * K_ + half * 8;
      const unsigned short* Bb = wcat + (long)(n0 + lm) * K_ + half * 8;
      v8f acc = {0.f, 0.f, 0.f, 0.f, 0.f, 0.f, 0.f, 0.f};
      for (int k0 = 0; k0 < K_; k0 += 32) {
        U16bf a, b;
        a.u[0] = *(const v4u*)(Ab + k0);
        a.u[1] = *(const v4u*)(Ab + k0 + 16);
        b.u[0] = *(const v4u*)(Bb + k0);
        b.u[1] = *(const v4u*)(Bb + k0 + 16);
        acc = __builtin_amdgcn_wmma_f32_16x16x32_bf16(
            false, a.v, false, b.v, (short)0, acc, false, false);
      }
      const float bn = bias[n0 + lm];
#pragma unroll
      for (int r = 0; r < 8; ++r) {
        int m = r + half * 8;                 // documented C/D layout
        smem[wave * 256 + m * 16 + lm] = acc[r] + bn;
      }
      __syncthreads();

      float hv2[2];
#pragma unroll
      for (int q = 0; q < 2; ++q) {
        int e = tid + q * 128;
        float iv = smem[e], fv = smem[256 + e], gv = smem[512 + e], ov = smem[768 + e];
        int b  = m0 + (e >> 4);
        int j  = j0g + (e & 15);
        float co = cbuf[b * D_ + j];
        float cn = sigm(fv) * co + sigm(iv) * tanh_f(gv);
        float hv = sigm(ov) * tanh_f(cn);
        cbuf[b * D_ + j] = cn;
        xbuf[(long)b * K_ + 2 * D_ + j] = f2bf(hv);   // h section for next GEMM
        hv2[q] = hv;
      }
      __syncthreads();
      smem[tid] = hv2[0];
      smem[tid + 128] = hv2[1];
      __syncthreads();
      // deterministic partial of phi = h @ W_g^T for this 16-col chunk of h
      for (int p = tid; p < 480; p += 128) {
        int bl = p / 30, r = p - bl * 30;
        const float* wg = W_g + (long)r * D_ + j0g;
        float s = 0.f;
#pragma unroll
        for (int kk = 0; kk < 16; ++kk) s += smem[bl * 16 + kk] * wg[kk];
        part[((long)(mtile * 64 + jtile) * 16 + bl) * 30 + r] = s;
      }
    }
    gsync(cnt, gen);

    // ---------- phase C: reduce phi, ksi/beta/alpha, attention weights w
    {
      const int b  = blk >> 2;       // each batch row handled by 4 blocks
      const int qq = blk & 3;
      const int mt = b >> 4, bl = b & 15;
      if (tid < 30) {
        float ph = b_g[tid];
        const float* pp = part + ((long)(mt * 64) * 16 + bl) * 30 + tid;
        for (int jt = 0; jt < 64; ++jt) ph += pp[(long)jt * 480];
        sphi[tid] = ph;
      }
      __syncthreads();
      const float* ksi_old = ksib + (t & 1) * (B_ * M_);
      float*       ksi_new = ksib + ((t + 1) & 1) * (B_ * M_);
      if (tid < 10) {
        float kn = ksi_old[b * M_ + tid] + __expf(sphi[tid]);
        sk[tid]  = kn;
        sbi[tid] = __expf(-sphi[10 + tid]);        // 1/beta
        if (qq == 0) ksi_new[b * M_ + tid] = kn;   // same value from the unique writer
      }
      if (tid == 0) {
        float amax = sphi[20];
        for (int m = 1; m < 10; ++m) amax = fmaxf(amax, sphi[20 + m]);
        float ex[10], asum = 0.f;
        for (int m = 0; m < 10; ++m) { ex[m] = __expf(sphi[20 + m] - amax); asum += ex[m]; }
        float inv = 1.f / asum;
        for (int m = 0; m < 10; ++m) sa[m] = ex[m] * inv;
      }
      __syncthreads();
      const int s = qq * 128 + tid;
      const float su = (float)s;
      float w = 0.f, t1 = 0.f;
#pragma unroll
      for (int m = 0; m < 10; ++m) {
        float a = sa[m], kn = sk[m], ib = sbi[m];
        float sR = sigm((su + 1.5f - kn) * ib);
        float sL = sigm((su + 0.5f - kn) * ib);
        w  += a * (sR - sL);
        t1 += a * sR;
      }
      wbuf[b * S_ + s] = w;
      out_align[((long)b * T_ + t) * S_ + s] = w;
      if (t == T_ - 1 && s == input_lengths[b] - 1) out_term[b] = 1.f - t1;
    }
    gsync(cnt, gen);

    // ---------- phase D: ctx GEMV (L2-resident memory), stage next x_t
    {
      int gid = blk * BLOCK_T + tid;           // 0..16383
      int item = gid * 2;                      // 2 adjacent d per lane
      int b  = item >> 10;
      int d0 = item & (D_ - 1);
      const float* mrow = memory + ((long)b * S_) * D_ + d0;
      const float* wr   = wbuf + b * S_;
      float a0 = 0.f, a1 = 0.f;
      for (int s = 0; s < S_; ++s) {
        float wv = wr[s];
        a0 += wv * mrow[0];
        a1 += wv * mrow[1];
        mrow += D_;
      }
      unsigned short* xb = xbuf + (long)b * K_;
      xb[D_ + d0]     = f2bf(a0);              // ctx section
      xb[D_ + d0 + 1] = f2bf(a1);
      if (t == T_ - 1) {
        out_ctx[b * D_ + d0]     = a0;
        out_ctx[b * D_ + d0 + 1] = a1;
      }
      if (t + 1 < T_) {                        // stage x_{t+1}
        const float* xin = input_h_c + ((long)b * T_ + (t + 1)) * D_ + d0;
        xb[d0]     = f2bf(xin[0]);
        xb[d0 + 1] = f2bf(xin[1]);
      }
    }
    gsync(cnt, gen);
  }
}

// ---------------- host ----------------
extern "C" void kernel_launch(void* const* d_in, const int* in_sizes, int n_in,
                              void* d_out, int out_size, void* d_ws, size_t ws_size,
                              hipStream_t stream) {
  (void)in_sizes; (void)n_in; (void)out_size; (void)ws_size;
  const float* input_h_c     = (const float*)d_in[0];
  const float* memory        = (const float*)d_in[1];
  const int*   input_lengths = (const int*)  d_in[2];
  const float* W_ih          = (const float*)d_in[3];
  const float* W_hh          = (const float*)d_in[4];
  const float* b_ih          = (const float*)d_in[5];
  const float* b_hh          = (const float*)d_in[6];
  const float* W_g           = (const float*)d_in[7];
  const float* b_g           = (const float*)d_in[8];

  char* ws = (char*)d_ws;
  constexpr size_t OFF_WCAT = 0;                                    // 4096*3072*2
  constexpr size_t OFF_BIAS = OFF_WCAT + (size_t)N4_ * K_ * 2;      // 4096*4
  constexpr size_t OFF_XBUF = OFF_BIAS + (size_t)N4_ * 4;           // 32*3072*2
  constexpr size_t OFF_CBUF = OFF_XBUF + (size_t)B_ * K_ * 2;       // 32*1024*4
  constexpr size_t OFF_KSI  = OFF_CBUF + (size_t)B_ * D_ * 4;       // 2*32*10*4
  constexpr size_t OFF_PART = OFF_KSI  + (size_t)2 * B_ * M_ * 4;   // 128*16*30*4
  constexpr size_t OFF_WBUF = OFF_PART + (size_t)128 * 16 * 30 * 4; // 32*512*4
  constexpr size_t OFF_SYNC = OFF_WBUF + (size_t)B_ * S_ * 4;

  unsigned short* wcat = (unsigned short*)(ws + OFF_WCAT);
  float*          bias = (float*)(ws + OFF_BIAS);
  unsigned short* xbuf = (unsigned short*)(ws + OFF_XBUF);
  float*          cbuf = (float*)(ws + OFF_CBUF);
  float*          ksib = (float*)(ws + OFF_KSI);
  float*          part = (float*)(ws + OFF_PART);
  float*          wbuf = (float*)(ws + OFF_WBUF);
  unsigned*       sync = (unsigned*)(ws + OFF_SYNC);

  float* out      = (float*)d_out;
  float* out_ctx  = out;                         // [B,1,D]
  float* out_algn = out + (size_t)B_ * D_;       // [B,T,S]
  float* out_term = out_algn + (size_t)B_ * T_ * S_;  // [B,1]

  {
    long n = (long)N4_ * K_;
    tts_init_wcat<<<(unsigned)((n + 255) / 256), 256, 0, stream>>>(W_ih, W_hh, wcat);
  }
  {
    int n = B_ * K_;
    tts_init_state<<<(n + 255) / 256, 256, 0, stream>>>(
        input_h_c, b_ih, b_hh, xbuf, cbuf, bias, ksib, sync);
  }
  tts_persistent<<<GRID_BLOCKS, BLOCK_T, 0, stream>>>(
      input_h_c, memory, input_lengths, W_g, b_g,
      wcat, bias, xbuf, cbuf, ksib, part, wbuf, sync,
      out_ctx, out_algn, out_term);
}
